// DynamicPatchSelection_41686952575477
// MI455X (gfx1250) — matrix operand order repaired and checked by reading, not compiled, not run on hardware
//
#include <hip/hip_runtime.h>

typedef __bf16 bf16_t;
typedef __attribute__((ext_vector_type(16))) __bf16 v16bf;
typedef __attribute__((ext_vector_type(8)))  float  v8f;

// ---------- bf16 <-> f32 (bit-exact) ----------
__device__ __forceinline__ bf16_t tobf(float f) {
    union { float f; unsigned u; } v; v.f = f;
    unsigned r = v.u + 0x7FFFu + ((v.u >> 16) & 1u);   // round-to-nearest-even
    unsigned short h = (unsigned short)(r >> 16);
    return __builtin_bit_cast(bf16_t, h);
}
__device__ __forceinline__ float tof(bf16_t b) {
    unsigned short h = __builtin_bit_cast(unsigned short, b);
    union { unsigned u; float f; } v; v.u = ((unsigned)h) << 16;
    return v.f;
}
__device__ __forceinline__ int iclamp(int v, int lo, int hi) {
    return v < lo ? lo : (v > hi ? hi : v);
}

// ---------- WMMA fragment helpers (CDNA5 16x16x32 bf16 layouts, ld=32) ----------
__device__ __forceinline__ v16bf frag_a(const bf16_t* As, int mrow0, int lane) {
    const bf16_t* p = As + (size_t)(mrow0 + (lane & 15)) * 32 + ((lane >> 4) << 3);
    v16bf a;
#pragma unroll
    for (int e = 0; e < 8; ++e) { a[e] = p[e]; a[e + 8] = p[e + 16]; }
    return a;
}
__device__ __forceinline__ v16bf frag_b(const bf16_t* Bt, int ncol0, int lane) {
    const bf16_t* p = Bt + (size_t)(ncol0 + (lane & 15)) * 32 + ((lane >> 4) << 3);
    v16bf b;
#pragma unroll
    for (int e = 0; e < 8; ++e) { b[e] = p[e]; b[e + 8] = p[e + 16]; }
    return b;
}
__device__ __forceinline__ v8f wmma_bf(v16bf a, v16bf b, v8f c) {
    return __builtin_amdgcn_wmma_f32_16x16x32_bf16(false, a, false, b, (short)0, c, false, false);
}

// ============================================================
// Weight prep
// ============================================================
__global__ void pad_convw_kernel(const float* __restrict__ cw, bf16_t* __restrict__ out) {
    int idx = blockIdx.x * 256 + threadIdx.x;      // 65536 total
    int p = idx >> 8, k = idx & 255;
    out[idx] = tobf(k < 243 ? cw[(size_t)p * 243 + k] : 0.f);
}

__global__ void transpose_bf_kernel(const float* __restrict__ W, bf16_t* __restrict__ Wt,
                                    int K, int N) {
    int n = blockIdx.y;
    int k = blockIdx.x * 256 + threadIdx.x;
    Wt[(size_t)n * K + k] = tobf(W[(size_t)k * N + n]);
}

// ============================================================
// Generic GEMM: C(MxN) = A_bf16(MxK) @ Bt_bf16(NxK)^T + bias -> bf16
// Macro tile 128x128, block 256 (8 waves: 2M x 4N), wave tile 64x32,
// double-buffered LDS, b128 staging, prefetch of tile s+2.
// All 6 fragment ds_loads hoisted ahead of the 8-WMMA burst.
// ============================================================
__global__ void gemm128_kernel(const bf16_t* __restrict__ A,
                               const bf16_t* __restrict__ Bt,
                               const float* __restrict__ bias,
                               bf16_t* __restrict__ C,
                               int M, int N, int K) {
    __shared__ __align__(16) bf16_t As[2][128 * 32];
    __shared__ __align__(16) bf16_t Bs[2][128 * 32];
    const int m0 = blockIdx.y * 128, n0 = blockIdx.x * 128;
    const int tid = threadIdx.x, lane = tid & 31, wave = tid >> 5;
    const int wm = (wave >> 2) * 64;    // 0 / 64
    const int wn = (wave & 3) * 32;     // 0..96
    v8f acc[4][2] = {};
    const int nsteps = K >> 5;

    auto stage = [&](int buf, int k0) {
#pragma unroll
        for (int t = 0; t < 2; ++t) {
            int idx = tid + t * 256;                 // 0..511
            int r = idx >> 2, seg = (idx & 3) << 3;  // 8 bf16 per chunk
            *(uint4*)&As[buf][r * 32 + seg] =
                *(const uint4*)&A[(size_t)(m0 + r) * K + k0 + seg];
            *(uint4*)&Bs[buf][r * 32 + seg] =
                *(const uint4*)&Bt[(size_t)(n0 + r) * K + k0 + seg];
        }
    };

    stage(0, 0);
    int cur = 0;
    for (int s = 0; s < nsteps; ++s) {
        __syncthreads();
        if (s + 1 < nsteps) stage(cur ^ 1, (s + 1) * 32);
        if (s + 2 < nsteps) {   // global_prefetch_b8 of tile s+2
            int r = tid >> 1;
            __builtin_prefetch(&A[(size_t)(m0 + r) * K + (s + 2) * 32], 0, 1);
            __builtin_prefetch(&Bt[(size_t)(n0 + r) * K + (s + 2) * 32], 0, 1);
        }
        const bf16_t* as = As[cur];
        const bf16_t* bs = Bs[cur];
        // hoist ALL fragment loads before the WMMA burst
        v16bf b0 = frag_b(bs, wn, lane);
        v16bf b1 = frag_b(bs, wn + 16, lane);
        v16bf a0 = frag_a(as, wm,      lane);
        v16bf a1 = frag_a(as, wm + 16, lane);
        v16bf a2 = frag_a(as, wm + 32, lane);
        v16bf a3 = frag_a(as, wm + 48, lane);
        acc[0][0] = wmma_bf(a0, b0, acc[0][0]);
        acc[0][1] = wmma_bf(a0, b1, acc[0][1]);
        acc[1][0] = wmma_bf(a1, b0, acc[1][0]);
        acc[1][1] = wmma_bf(a1, b1, acc[1][1]);
        acc[2][0] = wmma_bf(a2, b0, acc[2][0]);
        acc[2][1] = wmma_bf(a2, b1, acc[2][1]);
        acc[3][0] = wmma_bf(a3, b0, acc[3][0]);
        acc[3][1] = wmma_bf(a3, b1, acc[3][1]);
        cur ^= 1;
    }
    const int col = lane & 15, rb = (lane >> 4) * 8;
#pragma unroll
    for (int i = 0; i < 4; ++i)
#pragma unroll
        for (int j = 0; j < 2; ++j) {
            int nn = n0 + wn + j * 16 + col;
            float bv = bias[nn];
#pragma unroll
            for (int r = 0; r < 8; ++r) {
                int mm = m0 + wm + i * 16 + rb + r;
                C[(size_t)mm * N + nn] = tobf(acc[i][j][r] + bv);
            }
        }
}

// ============================================================
// Conv implicit GEMM: feat(b, P, HW) = relu(cwbf(256x256K) @ im2col + cb)
// grid: (HW/64, P/64, B), block 128 (4 waves 2x2), tile 64x64
// ============================================================
__global__ void conv_wmma_kernel(const float* __restrict__ x,
                                 const bf16_t* __restrict__ cwbf,
                                 const float* __restrict__ cb,
                                 bf16_t* __restrict__ featbf) {
    __shared__ __align__(16) bf16_t As[64 * 32];
    __shared__ __align__(16) bf16_t Bt[64 * 32];
    const int b  = blockIdx.z;
    const int m0 = blockIdx.y * 64;
    const int n0 = blockIdx.x * 64;
    const int tid = threadIdx.x, lane = tid & 31, wave = tid >> 5;
    const int wm = (wave >> 1) * 32, wn = (wave & 1) * 32;
    v8f acc[2][2] = {};
    for (int k0 = 0; k0 < 256; k0 += 32) {
        __syncthreads();
        for (int t = tid; t < 256; t += 128) {
            int r = t >> 2, seg = (t & 3) << 3;
            *(uint4*)&As[r * 32 + seg] =
                *(const uint4*)&cwbf[(size_t)(m0 + r) * 256 + k0 + seg];
        }
        for (int idx = tid; idx < 64 * 32; idx += 128) {
            int n = idx >> 5, kk = idx & 31, k = k0 + kk;
            float v = 0.f;
            if (k < 243) {
                int c = k / 81, rem = k % 81, ky = rem / 9, kx = rem % 9;
                int ng = n0 + n;
                int oy = ng >> 6, ox = ng & 63;
                int iy = oy + ky - 4, ix = ox + kx - 4;
                if (iy >= 0 && iy < 64 && ix >= 0 && ix < 64)
                    v = x[(((size_t)b * 3 + c) * 64 + iy) * 64 + ix];
            }
            Bt[idx] = tobf(v);
        }
        __syncthreads();
        v16bf b0 = frag_b(Bt, wn, lane);
        v16bf b1 = frag_b(Bt, wn + 16, lane);
        v16bf a0 = frag_a(As, wm, lane);
        v16bf a1 = frag_a(As, wm + 16, lane);
        acc[0][0] = wmma_bf(a0, b0, acc[0][0]);
        acc[0][1] = wmma_bf(a0, b1, acc[0][1]);
        acc[1][0] = wmma_bf(a1, b0, acc[1][0]);
        acc[1][1] = wmma_bf(a1, b1, acc[1][1]);
    }
    const int col = lane & 15, rb = (lane >> 4) * 8;
#pragma unroll
    for (int i = 0; i < 2; ++i)
#pragma unroll
        for (int j = 0; j < 2; ++j) {
            int nn = n0 + wn + j * 16 + col;
#pragma unroll
            for (int r = 0; r < 8; ++r) {
                int mm = m0 + wm + i * 16 + rb + r;
                float v = fmaxf(acc[i][j][r] + cb[mm], 0.f);
                featbf[((size_t)b * 256 + mm) * 4096 + nn] = tobf(v);
            }
        }
}

// ============================================================
// Attention scores: scores[z] = (q_z @ k_z^T)/sqrt(32), z=(b,h); K=32 one step
// grid: (4, 4, 128), block 128
// ============================================================
__global__ void attn_scores_kernel(const bf16_t* __restrict__ q,
                                   const bf16_t* __restrict__ k,
                                   float* __restrict__ scores) {
    __shared__ __align__(16) bf16_t As[64 * 32];
    __shared__ __align__(16) bf16_t Bt[64 * 32];
    const int z = blockIdx.z, b = z >> 3, h = z & 7;
    const bf16_t* Q  = q + (size_t)b * 256 * 256 + h * 32;
    const bf16_t* Kp = k + (size_t)b * 256 * 256 + h * 32;
    const int m0 = blockIdx.y * 64, n0 = blockIdx.x * 64;
    const int tid = threadIdx.x, lane = tid & 31, wave = tid >> 5;
    const int wm = (wave >> 1) * 32, wn = (wave & 1) * 32;
    for (int t = tid; t < 256; t += 128) {
        int r = t >> 2, seg = (t & 3) << 3;
        *(uint4*)&As[r * 32 + seg] = *(const uint4*)&Q [(size_t)(m0 + r) * 256 + seg];
        *(uint4*)&Bt[r * 32 + seg] = *(const uint4*)&Kp[(size_t)(n0 + r) * 256 + seg];
    }
    __syncthreads();
    v8f acc[2][2] = {};
    v16bf b0 = frag_b(Bt, wn, lane);
    v16bf b1 = frag_b(Bt, wn + 16, lane);
    v16bf a0 = frag_a(As, wm, lane);
    v16bf a1 = frag_a(As, wm + 16, lane);
    acc[0][0] = wmma_bf(a0, b0, acc[0][0]);
    acc[0][1] = wmma_bf(a0, b1, acc[0][1]);
    acc[1][0] = wmma_bf(a1, b0, acc[1][0]);
    acc[1][1] = wmma_bf(a1, b1, acc[1][1]);
    const float scale = 0.17677669529663687f;   // 1/sqrt(32)
    const int col = lane & 15, rb = (lane >> 4) * 8;
    float* S = scores + (size_t)z * 256 * 256;
#pragma unroll
    for (int i = 0; i < 2; ++i)
#pragma unroll
        for (int j = 0; j < 2; ++j) {
            int nn = n0 + wn + j * 16 + col;
#pragma unroll
            for (int r = 0; r < 8; ++r) {
                int mm = m0 + wm + i * 16 + rb + r;
                S[(size_t)mm * 256 + nn] = acc[i][j][r] * scale;
            }
        }
}

// ============================================================
// Row softmax in place; one wave per 256-wide row, contiguous 8/lane
// ============================================================
__global__ void softmax_kernel(float* __restrict__ scores) {
    const int row  = blockIdx.x * 8 + (threadIdx.x >> 5);
    const int lane = threadIdx.x & 31;
    float* p = scores + (size_t)row * 256 + lane * 8;
    float v[8];
    *(float4*)&v[0] = *(const float4*)&p[0];
    *(float4*)&v[4] = *(const float4*)&p[4];
    float mx = -3.0e38f;
#pragma unroll
    for (int e = 0; e < 8; ++e) mx = fmaxf(mx, v[e]);
#pragma unroll
    for (int off = 16; off > 0; off >>= 1) mx = fmaxf(mx, __shfl_xor(mx, off));
    float sum = 0.f;
#pragma unroll
    for (int e = 0; e < 8; ++e) { v[e] = __expf(v[e] - mx); sum += v[e]; }
#pragma unroll
    for (int off = 16; off > 0; off >>= 1) sum += __shfl_xor(sum, off);
    float inv = 1.f / sum;
#pragma unroll
    for (int e = 0; e < 8; ++e) v[e] *= inv;
    *(float4*)&p[0] = *(const float4*)&v[0];
    *(float4*)&p[4] = *(const float4*)&v[4];
}

// ============================================================
// o = attn @ v per (b,h); tile 128x32, block 128 (4 waves stacked in M)
// ============================================================
__global__ void attn_v_kernel(const float* __restrict__ scores,
                              const bf16_t* __restrict__ v,
                              bf16_t* __restrict__ o) {
    __shared__ __align__(16) bf16_t As[128 * 32];
    __shared__ __align__(16) bf16_t Bt[32 * 32];
    const int z = blockIdx.z, b = z >> 3, h = z & 7;
    const float* Aattn = scores + (size_t)z * 256 * 256;
    const bf16_t* V = v + (size_t)b * 256 * 256 + h * 32;
    const int m0 = blockIdx.y * 128;
    const int tid = threadIdx.x, lane = tid & 31, wave = tid >> 5;
    v8f acc[2][2] = {};
    for (int k0 = 0; k0 < 256; k0 += 32) {
        __syncthreads();
        for (int t = tid; t < 512; t += 128) {
            int r = t >> 2, seg = (t & 3) << 3;
            const float* src = &Aattn[(size_t)(m0 + r) * 256 + k0 + seg];
            float4 f0 = *(const float4*)src;
            float4 f1 = *(const float4*)(src + 4);
            union { bf16_t h[8]; uint4 u; } pk;
            pk.h[0] = tobf(f0.x); pk.h[1] = tobf(f0.y); pk.h[2] = tobf(f0.z); pk.h[3] = tobf(f0.w);
            pk.h[4] = tobf(f1.x); pk.h[5] = tobf(f1.y); pk.h[6] = tobf(f1.z); pk.h[7] = tobf(f1.w);
            *(uint4*)&As[r * 32 + seg] = pk.u;
        }
        for (int idx = tid; idx < 32 * 32; idx += 128) {
            int n = idx >> 5, kk = idx & 31;
            Bt[idx] = V[(size_t)(k0 + kk) * 256 + n];
        }
        __syncthreads();
        const int wm = wave * 32;
        v16bf b0 = frag_b(Bt, 0, lane);
        v16bf b1 = frag_b(Bt, 16, lane);
        v16bf a0 = frag_a(As, wm, lane);
        v16bf a1 = frag_a(As, wm + 16, lane);
        acc[0][0] = wmma_bf(a0, b0, acc[0][0]);
        acc[0][1] = wmma_bf(a0, b1, acc[0][1]);
        acc[1][0] = wmma_bf(a1, b0, acc[1][0]);
        acc[1][1] = wmma_bf(a1, b1, acc[1][1]);
    }
    const int col = lane & 15, rb = (lane >> 4) * 8;
#pragma unroll
    for (int i = 0; i < 2; ++i)
#pragma unroll
        for (int j = 0; j < 2; ++j) {
            int nn = j * 16 + col;
#pragma unroll
            for (int r = 0; r < 8; ++r) {
                int mm = m0 + wave * 32 + i * 16 + rb + r;
                o[((size_t)b * 256 + mm) * 256 + h * 32 + nn] = tobf(acc[i][j][r]);
            }
        }
}

// ============================================================
// tp = tanh(feat2 @ fc_w + fc_b); one block per (b,p) row
// ============================================================
__global__ void fc_tanh_kernel(const bf16_t* __restrict__ feat2,
                               const float* __restrict__ fc_w,
                               const float* __restrict__ fc_b,
                               float* __restrict__ tp) {
    __shared__ float s0[256];
    __shared__ float s1[256];
    const int row = blockIdx.x, tid = threadIdx.x;
    const bf16_t* p = feat2 + (size_t)row * 4096 + tid * 16;
    union { bf16_t h[16]; uint4 u[2]; } pk;
    pk.u[0] = *(const uint4*)&p[0];
    pk.u[1] = *(const uint4*)&p[8];
    float a0 = 0.f, a1 = 0.f;
    const float* w = fc_w + (size_t)tid * 32;
#pragma unroll
    for (int e = 0; e < 16; ++e) {
        float v = tof(pk.h[e]);
        a0 += v * w[e * 2];
        a1 += v * w[e * 2 + 1];
    }
    s0[tid] = a0; s1[tid] = a1;
    __syncthreads();
    for (int s = 128; s > 0; s >>= 1) {
        if (tid < s) { s0[tid] += s0[tid + s]; s1[tid] += s1[tid + s]; }
        __syncthreads();
    }
    if (tid == 0) {
        tp[(size_t)row * 2]     = tanhf(s0[0] + fc_b[0]);
        tp[(size_t)row * 2 + 1] = tanhf(s1[0] + fc_b[1]);
    }
}

// ============================================================
// Bilinear patch sampling + pos embedding -> d_out
// ============================================================
__global__ void sample_pos_kernel(const float* __restrict__ x,
                                  const float* __restrict__ tp,
                                  const float* __restrict__ pos_w,
                                  const float* __restrict__ pos_b,
                                  float* __restrict__ out_patches,
                                  float* __restrict__ out_pos) {
    const int bp = blockIdx.x;
    const int b  = bp >> 8;
    const int tid = threadIdx.x;
    const float tx = tp[(size_t)bp * 2];
    const float ty = tp[(size_t)bp * 2 + 1];
    if (tid < 81) {
        int i = tid / 9, j = tid % 9;
        float basej = (2.f * j + 1.f) / 9.f - 1.f;
        float basei = (2.f * i + 1.f) / 9.f - 1.f;
        float gx = (9.f / 64.f) * basej + tx;
        float gy = (9.f / 64.f) * basei + ty;
        float ix = ((gx + 1.f) * 64.f - 1.f) * 0.5f;
        float iy = ((gy + 1.f) * 64.f - 1.f) * 0.5f;
        float x0f = floorf(ix), y0f = floorf(iy);
        float wx = ix - x0f, wy = iy - y0f;
        int x0i = iclamp((int)x0f, 0, 63);
        int x1i = iclamp((int)x0f + 1, 0, 63);
        int y0i = iclamp((int)y0f, 0, 63);
        int y1i = iclamp((int)y0f + 1, 0, 63);
        float w00 = (1.f - wx) * (1.f - wy), w01 = wx * (1.f - wy);
        float w10 = (1.f - wx) * wy,         w11 = wx * wy;
#pragma unroll
        for (int c = 0; c < 3; ++c) {
            const float* img = x + (((size_t)b * 3 + c) * 64) * 64;
            float v = img[y0i * 64 + x0i] * w00 + img[y0i * 64 + x1i] * w01 +
                      img[y1i * 64 + x0i] * w10 + img[y1i * 64 + x1i] * w11;
            out_patches[(size_t)bp * 243 + c * 81 + tid] = v;
        }
    } else if (tid >= 96 && tid < 160) {
        int d = tid - 96;
        out_pos[(size_t)bp * 64 + d] = tx * pos_w[d] + ty * pos_w[64 + d] + pos_b[d];
    }
}

// ============================================================
// Host launcher
// ============================================================
extern "C" void kernel_launch(void* const* d_in, const int* in_sizes, int n_in,
                              void* d_out, int out_size, void* d_ws, size_t ws_size,
                              hipStream_t stream) {
    const float* x      = (const float*)d_in[0];
    const float* conv_w = (const float*)d_in[1];
    const float* conv_b = (const float*)d_in[2];
    const float* in_w   = (const float*)d_in[3];
    const float* in_b   = (const float*)d_in[4];
    const float* wq     = (const float*)d_in[5];
    const float* bq     = (const float*)d_in[6];
    const float* wk     = (const float*)d_in[7];
    const float* bk     = (const float*)d_in[8];
    const float* wv     = (const float*)d_in[9];
    const float* bv     = (const float*)d_in[10];
    const float* wo     = (const float*)d_in[11];
    const float* bo     = (const float*)d_in[12];
    const float* out_w  = (const float*)d_in[13];
    const float* out_b  = (const float*)d_in[14];
    const float* fc_w   = (const float*)d_in[15];
    const float* fc_b   = (const float*)d_in[16];
    const float* pos_w  = (const float*)d_in[17];
    const float* pos_b  = (const float*)d_in[18];

    char* ws = (char*)d_ws;
    size_t off = 0;
    auto alloc = [&](size_t bytes) { void* p = ws + off; off += (bytes + 255) & ~(size_t)255; return p; };
    bf16_t* featbf  = (bf16_t*)alloc((size_t)4096 * 4096 * 2);
    bf16_t* tokbf   = (bf16_t*)alloc((size_t)4096 * 256 * 2);
    bf16_t* qbf     = (bf16_t*)alloc((size_t)4096 * 256 * 2);
    bf16_t* kbf     = (bf16_t*)alloc((size_t)4096 * 256 * 2);
    bf16_t* vbf     = (bf16_t*)alloc((size_t)4096 * 256 * 2);
    float*  scoresf = (float*) alloc((size_t)128 * 256 * 256 * 4);
    bf16_t* obf     = (bf16_t*)alloc((size_t)4096 * 256 * 2);
    bf16_t* o2bf    = (bf16_t*)alloc((size_t)4096 * 256 * 2);
    bf16_t* feat2bf = (bf16_t*)alloc((size_t)4096 * 4096 * 2);
    float*  tpf     = (float*) alloc((size_t)4096 * 2 * 4);
    bf16_t* cwbf    = (bf16_t*)alloc((size_t)256 * 256 * 2);
    bf16_t* inw_t   = (bf16_t*)alloc((size_t)256 * 4096 * 2);
    bf16_t* wq_t    = (bf16_t*)alloc((size_t)256 * 256 * 2);
    bf16_t* wk_t    = (bf16_t*)alloc((size_t)256 * 256 * 2);
    bf16_t* wv_t    = (bf16_t*)alloc((size_t)256 * 256 * 2);
    bf16_t* wo_t    = (bf16_t*)alloc((size_t)256 * 256 * 2);
    bf16_t* outw_t  = (bf16_t*)alloc((size_t)4096 * 256 * 2);
    (void)ws_size; (void)in_sizes; (void)n_in; (void)out_size;

    pad_convw_kernel<<<dim3(256), 256, 0, stream>>>(conv_w, cwbf);
    transpose_bf_kernel<<<dim3(16, 256), 256, 0, stream>>>(in_w,  inw_t,  4096, 256);
    transpose_bf_kernel<<<dim3(1, 256),  256, 0, stream>>>(wq,    wq_t,   256,  256);
    transpose_bf_kernel<<<dim3(1, 256),  256, 0, stream>>>(wk,    wk_t,   256,  256);
    transpose_bf_kernel<<<dim3(1, 256),  256, 0, stream>>>(wv,    wv_t,   256,  256);
    transpose_bf_kernel<<<dim3(1, 256),  256, 0, stream>>>(wo,    wo_t,   256,  256);
    transpose_bf_kernel<<<dim3(1, 4096), 256, 0, stream>>>(out_w, outw_t, 256,  4096);

    conv_wmma_kernel<<<dim3(64, 4, 16), 128, 0, stream>>>(x, cwbf, conv_b, featbf);
    gemm128_kernel<<<dim3(2, 32), 256, 0, stream>>>(featbf, inw_t, in_b, tokbf, 4096, 256, 4096);
    gemm128_kernel<<<dim3(2, 32), 256, 0, stream>>>(tokbf, wq_t, bq, qbf, 4096, 256, 256);
    gemm128_kernel<<<dim3(2, 32), 256, 0, stream>>>(tokbf, wk_t, bk, kbf, 4096, 256, 256);
    gemm128_kernel<<<dim3(2, 32), 256, 0, stream>>>(tokbf, wv_t, bv, vbf, 4096, 256, 256);
    attn_scores_kernel<<<dim3(4, 4, 128), 128, 0, stream>>>(qbf, kbf, scoresf);
    softmax_kernel<<<dim3(4096), 256, 0, stream>>>(scoresf);
    attn_v_kernel<<<dim3(1, 2, 128), 128, 0, stream>>>(scoresf, vbf, obf);
    gemm128_kernel<<<dim3(2, 32), 256, 0, stream>>>(obf, wo_t, bo, o2bf, 4096, 256, 256);
    gemm128_kernel<<<dim3(32, 32), 256, 0, stream>>>(o2bf, outw_t, out_b, feat2bf, 4096, 4096, 256);
    fc_tanh_kernel<<<dim3(4096), 256, 0, stream>>>(feat2bf, fc_w, fc_b, tpf);
    float* out_patches = (float*)d_out;
    float* out_pos     = (float*)d_out + (size_t)4096 * 243;
    sample_pos_kernel<<<dim3(4096), 160, 0, stream>>>(x, tpf, pos_w, pos_b, out_patches, out_pos);
}
